// Attention_40114994544760
// MI455X (gfx1250) — compile-verified, matrix operand order checked by reading, not run on hardware
//
#include <hip/hip_runtime.h>
#include <hip/hip_bf16.h>

typedef __attribute__((ext_vector_type(16))) _Float16 v16h;
typedef __attribute__((ext_vector_type(8)))  _Float16 v8h;
typedef __attribute__((ext_vector_type(8)))  float    v8f;

#define B_    4
#define Q_    256
#define H_    32
#define D_    128
#define KVH_  8
#define G_    4
#define BLOCK_ 16
#define NB_   128
#define KV_   2048
#define SCALE_ 0.08838834764831845f  /* 1/sqrt(128) */

__device__ __forceinline__ float rmax16(float v) {
  v = fmaxf(v, __shfl_xor(v, 1, 32));
  v = fmaxf(v, __shfl_xor(v, 2, 32));
  v = fmaxf(v, __shfl_xor(v, 4, 32));
  v = fmaxf(v, __shfl_xor(v, 8, 32));
  return v;
}
__device__ __forceinline__ float rsum16(float v) {
  v += __shfl_xor(v, 1, 32);
  v += __shfl_xor(v, 2, 32);
  v += __shfl_xor(v, 4, 32);
  v += __shfl_xor(v, 8, 32);
  return v;
}
__device__ __forceinline__ v16h cat8(v8h lo, v8h hi) {
  return __builtin_shufflevector(lo, hi, 0, 1, 2, 3, 4, 5, 6, 7,
                                 8, 9, 10, 11, 12, 13, 14, 15);
}

// 4 waves per workgroup (one per GQA group g) share one (b, kv-head, q-tile).
// K/V chunks are staged once into double-buffered LDS (f16); staging of chunk
// n+1 overlaps compute of chunk n, one barrier per chunk.
__global__ __launch_bounds__(128)
void paged_gqa_prefill_wmma(const float* __restrict__ q,
                            const float* __restrict__ kc,
                            const float* __restrict__ vc,
                            const int*   __restrict__ bt,
                            const int*   __restrict__ slens,
                            float* __restrict__ out) {
  __shared__ _Float16 ldsK[2][32 * D_];   // [buf][token][d]       (2x8 KB)
  __shared__ _Float16 ldsV[2][D_ * 32];   // [buf][d][token]       (2x8 KB)
  __shared__ _Float16 ldsP[G_][16 * 32];  // per-wave P tile       (4 KB)

  const int tid  = threadIdx.x;
  const int w    = tid >> 5;           // wave id == query-head group g
  const int lane = tid & 31;
  const int l16  = lane & 15;
  const int hi   = lane >> 4;          // lane half
  const int base = hi * 8;             // A/B fragment k-base (ISA 7.12.2)
  const int hi8  = hi * 8;             // C/D row base

  int bid = blockIdx.x;
  const int qt  = bid & 15;  bid >>= 4;
  const int kvh = bid & 7;   bid >>= 3;
  const int b   = bid;
  const int h   = kvh * G_ + w;
  const int q0  = qt * 16;

  int seq = slens[b]; if (seq < Q_) seq = Q_;
  const int sbq     = seq - Q_ + q0;          // q_pos of tile row 0
  const int nchunks = (sbq + 16 + 31) >> 5;   // 32-kv chunks needed (causal)

  const int* btab = bt + b * NB_;

  // staging role: token = tid&31, d-quarter = tid>>5 (32 d values each)
  const int st_t  = tid & 31;
  const int st_dq = tid >> 5;

  auto stage = [&](int c, int buf) {
    _Float16* sK = ldsK[buf];
    _Float16* sV = ldsV[buf];
    const int t  = c * 32 + st_t;
    const int pb = btab[t >> 4];
    const size_t rowoff =
        (size_t)((pb * BLOCK_ + (t & 15)) * KVH_ + kvh) * D_ + st_dq * 32;
    const float* krow = kc + rowoff;
    const float* vrow = vc + rowoff;

    if (c + 1 < nchunks) {  // prefetch one more chunk ahead (global_prefetch_b8)
      const int tn  = c * 32 + 32 + st_t;
      const int pbn = btab[tn >> 4];
      const size_t noff =
          (size_t)((pbn * BLOCK_ + (tn & 15)) * KVH_ + kvh) * D_ + st_dq * 32;
      __builtin_prefetch(kc + noff, 0, 3);
      __builtin_prefetch(vc + noff, 0, 3);
    }

    #pragma unroll
    for (int j8 = 0; j8 < 4; ++j8) {
      float4 ka = *(const float4*)(krow + j8 * 8);
      float4 kb = *(const float4*)(krow + j8 * 8 + 4);
      v8h kp;
      kp[0] = (_Float16)ka.x; kp[1] = (_Float16)ka.y;
      kp[2] = (_Float16)ka.z; kp[3] = (_Float16)ka.w;
      kp[4] = (_Float16)kb.x; kp[5] = (_Float16)kb.y;
      kp[6] = (_Float16)kb.z; kp[7] = (_Float16)kb.w;
      *(v8h*)(&sK[st_t * D_ + st_dq * 32 + j8 * 8]) = kp;

      float4 va = *(const float4*)(vrow + j8 * 8);
      float4 vb = *(const float4*)(vrow + j8 * 8 + 4);
      const int d0 = st_dq * 32 + j8 * 8;
      sV[(d0 + 0) * 32 + st_t] = (_Float16)va.x;
      sV[(d0 + 1) * 32 + st_t] = (_Float16)va.y;
      sV[(d0 + 2) * 32 + st_t] = (_Float16)va.z;
      sV[(d0 + 3) * 32 + st_t] = (_Float16)va.w;
      sV[(d0 + 4) * 32 + st_t] = (_Float16)vb.x;
      sV[(d0 + 5) * 32 + st_t] = (_Float16)vb.y;
      sV[(d0 + 6) * 32 + st_t] = (_Float16)vb.z;
      sV[(d0 + 7) * 32 + st_t] = (_Float16)vb.w;
    }
  };

  // ---- Q tile [16 x 128] -> 4 A-fragments (f16, pre-scaled) ----
  v16h qf[4];
  {
    const float* qrow = q + (size_t)((b * Q_ + q0 + l16) * H_ + h) * D_;
    #pragma unroll
    for (int dc = 0; dc < 4; ++dc) {
      #pragma unroll
      for (int j = 0; j < 8; ++j) {
        qf[dc][j]     = (_Float16)(qrow[dc * 32 + base + j]      * SCALE_);
        qf[dc][8 + j] = (_Float16)(qrow[dc * 32 + base + 16 + j] * SCALE_);
      }
    }
  }

  v8f zero = {};
  v8f o[8];
  #pragma unroll
  for (int nt = 0; nt < 8; ++nt) o[nt] = zero;
  float mrow[8], lrow[8];
  #pragma unroll
  for (int r = 0; r < 8; ++r) { mrow[r] = -1e30f; lrow[r] = 0.f; }

  stage(0, 0);
  __syncthreads();

  for (int cb = 0; cb < nchunks; ++cb) {
    const int cur = cb & 1;
    const int kv0 = cb * 32;

    // overlap: stage next chunk into the spare buffer while computing this one
    if (cb + 1 < nchunks) stage(cb + 1, cur ^ 1);

    // ---- QK^T: preload all 8 K fragments, then 8 dense WMMAs ----
    v8f s0 = {}, s1 = {};
    {
      const _Float16* bK = ldsK[cur];
      const int t0row = l16 * D_;          // B-frag: lane = token column
      const int t1row = t0row + 16 * D_;
      v16h kfa[4], kfb[4];
      #pragma unroll
      for (int dc = 0; dc < 4; ++dc) {
        v8h a0 = *(const v8h*)(&bK[t0row + dc * 32 + base]);
        v8h a1 = *(const v8h*)(&bK[t0row + dc * 32 + base + 16]);
        v8h b0 = *(const v8h*)(&bK[t1row + dc * 32 + base]);
        v8h b1 = *(const v8h*)(&bK[t1row + dc * 32 + base + 16]);
        kfa[dc] = cat8(a0, a1);
        kfb[dc] = cat8(b0, b1);
      }
      #pragma unroll
      for (int dc = 0; dc < 4; ++dc) {
        s0 = __builtin_amdgcn_wmma_f32_16x16x32_f16(false, qf[dc], false,
                                                    kfa[dc], (short)0, s0,
                                                    false, false);
        s1 = __builtin_amdgcn_wmma_f32_16x16x32_f16(false, qf[dc], false,
                                                    kfb[dc], (short)0, s1,
                                                    false, false);
      }
    }

    // ---- masked online softmax (rows live in C-layout: m = r + hi8) ----
    float alpha[8];
    #pragma unroll
    for (int r = 0; r < 8; ++r) {
      const int m  = r + hi8;
      const int qp = sbq + m;                       // global query position
      const bool v0 = (kv0 + l16)      <= qp;       // causal mask
      const bool v1 = (kv0 + 16 + l16) <= qp;
      float a0 = v0 ? s0[r] : -1e30f;
      float a1 = v1 ? s1[r] : -1e30f;
      float rowmax = rmax16(fmaxf(a0, a1));
      float mnew   = fmaxf(mrow[r], rowmax);
      alpha[r] = __expf(mrow[r] - mnew);
      float p0 = v0 ? __expf(a0 - mnew) : 0.f;
      float p1 = v1 ? __expf(a1 - mnew) : 0.f;
      lrow[r] = lrow[r] * alpha[r] + rsum16(p0 + p1);
      mrow[r] = mnew;
      ldsP[w][m * 32 + l16]      = (_Float16)p0;    // A-layout relayout
      ldsP[w][m * 32 + 16 + l16] = (_Float16)p1;
    }

    // ---- O = O*alpha + P @ V (V fragments preloaded in groups of 4) ----
    v16h pf;
    {
      v8h plo = *(const v8h*)(&ldsP[w][l16 * 32 + base]);
      v8h phi = *(const v8h*)(&ldsP[w][l16 * 32 + base + 16]);
      pf = cat8(plo, phi);
    }
    {
      const _Float16* bV = ldsV[cur];
      #pragma unroll
      for (int grp = 0; grp < 2; ++grp) {
        v16h vf[4];
        #pragma unroll
        for (int i = 0; i < 4; ++i) {
          const int d = (grp * 4 + i) * 16 + l16;
          v8h vlo = *(const v8h*)(&bV[d * 32 + base]);
          v8h vhi = *(const v8h*)(&bV[d * 32 + base + 16]);
          vf[i] = cat8(vlo, vhi);
        }
        #pragma unroll
        for (int i = 0; i < 4; ++i) {
          const int nt = grp * 4 + i;
          v8f oc = o[nt];
          #pragma unroll
          for (int r = 0; r < 8; ++r) oc[r] *= alpha[r];
          o[nt] = __builtin_amdgcn_wmma_f32_16x16x32_f16(false, pf, false,
                                                         vf[i], (short)0, oc,
                                                         false, false);
        }
      }
    }

    __syncthreads();  // staging of next buffer done; current buffer released
  }

  // ---- epilogue: divide by l, write coalesced f32 rows ----
  #pragma unroll
  for (int r = 0; r < 8; ++r) {
    const int m    = r + hi8;
    const float iv = 1.0f / lrow[r];
    float* orow = out + (size_t)((b * Q_ + q0 + m) * H_ + h) * D_;
    #pragma unroll
    for (int nt = 0; nt < 8; ++nt)
      orow[nt * 16 + l16] = o[nt][r] * iv;
  }
}

extern "C" void kernel_launch(void* const* d_in, const int* in_sizes, int n_in,
                              void* d_out, int out_size, void* d_ws, size_t ws_size,
                              hipStream_t stream) {
  (void)in_sizes; (void)n_in; (void)out_size; (void)d_ws; (void)ws_size;
  const float* query        = (const float*)d_in[0];
  const float* k_cache      = (const float*)d_in[1];
  const float* v_cache      = (const float*)d_in[2];
  const int*   block_tables = (const int*)d_in[3];
  const int*   seq_lens     = (const int*)d_in[4];
  float*       outp         = (float*)d_out;

  // B * KVH * (Q/16) = 4*8*16 = 512 workgroups of 4 wave32s (one per g)
  dim3 grid(B_ * KVH_ * (Q_ / 16));
  dim3 block(128);
  paged_gqa_prefill_wmma<<<grid, block, 0, stream>>>(
      query, k_cache, v_cache, block_tables, seq_lens, outp);
}
